// Attention_20736102105614
// MI455X (gfx1250) — compile-verified, matrix operand order checked by reading, not run on hardware
//
#include <hip/hip_runtime.h>
#include <hip/hip_bf16.h>

typedef __attribute__((ext_vector_type(16))) __bf16 v16bf;
typedef __attribute__((ext_vector_type(8)))  __bf16 bf16x8;
typedef __attribute__((ext_vector_type(8)))  float  v8f;
typedef int v4i_gcc __attribute__((vector_size(16)));

#define AS1 __attribute__((address_space(1)))
#define AS3 __attribute__((address_space(3)))

#if __has_builtin(__builtin_amdgcn_global_load_async_to_lds_b128)
#define HAVE_ASYNC_LDS 1
#else
#define HAVE_ASYNC_LDS 0
#endif

static __device__ __forceinline__ v8f wmma_bf16(v16bf a, v16bf b, v8f c) {
    // D = A(16x32 bf16) * B(32x16 bf16) + C(16x16 f32)
    return __builtin_amdgcn_wmma_f32_16x16x32_bf16(
        /*neg_a=*/false, a, /*neg_b=*/false, b,
        /*c_mod=*/(short)0, c, /*reuse_a=*/false, /*reuse_b=*/false);
}

// ---------------------------------------------------------------------------
// fp32 -> bf16 conversion
// ---------------------------------------------------------------------------
__global__ void cvt_f32_bf16_kernel(const float* __restrict__ in,
                                    __bf16* __restrict__ out, int n) {
    int i = blockIdx.x * blockDim.x + threadIdx.x;
    if (i < n) out[i] = (__bf16)in[i];
}

// ---------------------------------------------------------------------------
// Generic bf16 WMMA GEMM:  C[m,n] = scale * sum_k A[m,k] * B[n,k]
//   A: [M,K] row-major bf16,  B: [N,K] row-major bf16 (i.e. y = A @ B^T)
// mode 0: bf16 out, token-major [M,N]
// mode 1: bf16 out, head-major  [B, H, T, 64]   (m = b*T+t, n = h*64+d)
// mode 2: f32  out, token-major [M,N]
// One wave computes a 16(m) x 64(n) strip; 8 waves / block.
// Ping-pong double buffering with FIXED buffer sets (no register rotation):
// each buffer is consumed by WMMAs and then reloaded in place, so the
// compiler needs no v_mov shuffles and loads stay 2 chunks in flight.
// Requires K % 64 == 0 and K >= 64 (K = 1024 here).
// ---------------------------------------------------------------------------
__global__ __launch_bounds__(256)
void gemm_bf16_kernel(const __bf16* __restrict__ A, const __bf16* __restrict__ Bm,
                      void* __restrict__ out, int M, int N, int K,
                      int mode, float scale, int Tdim, int Hheads) {
    const int lane = threadIdx.x & 31;
    const int wave = threadIdx.x >> 5;
    const int gw   = blockIdx.x * (blockDim.x >> 5) + wave;
    const int mTiles = M >> 4, nStrips = N >> 6;
    if (gw >= mTiles * nStrips) return;
    const int wm = gw % mTiles, wn = gw / mTiles;
    const int l16 = lane & 15, hi = lane >> 4;

    const int rowA = wm * 16 + l16;
    const __bf16* Arow = A + (size_t)rowA * K + hi * 16;
    const __bf16* Brow[4];
#pragma unroll
    for (int t = 0; t < 4; ++t) {
        int col = wn * 64 + t * 16 + l16;
        Brow[t] = Bm + (size_t)col * K + hi * 16;
    }

    v8f acc[4] = {v8f{}, v8f{}, v8f{}, v8f{}};

    // prime both buffers: buf A <- chunk 0, buf B <- chunk 1
    v16bf aA = *(const v16bf*)(Arow);
    v16bf bA[4];
#pragma unroll
    for (int t = 0; t < 4; ++t) bA[t] = *(const v16bf*)(Brow[t]);
    v16bf aB = *(const v16bf*)(Arow + 32);
    v16bf bB[4];
#pragma unroll
    for (int t = 0; t < 4; ++t) bB[t] = *(const v16bf*)(Brow[t] + 32);

    for (int k = 64; k < K; k += 64) {
        // speculative prefetch ~8 chunks ahead (dropped if invalid)
        __builtin_prefetch(Arow + k + 256, 0, 3);
#pragma unroll
        for (int t = 0; t < 4; ++t) __builtin_prefetch(Brow[t] + k + 256, 0, 3);

        // consume buffer A (chunk k-64), reload in place with chunk k
#pragma unroll
        for (int t = 0; t < 4; ++t) acc[t] = wmma_bf16(aA, bA[t], acc[t]);
        aA = *(const v16bf*)(Arow + k);
#pragma unroll
        for (int t = 0; t < 4; ++t) bA[t] = *(const v16bf*)(Brow[t] + k);

        // consume buffer B (chunk k-32), reload in place with chunk k+32
#pragma unroll
        for (int t = 0; t < 4; ++t) acc[t] = wmma_bf16(aB, bB[t], acc[t]);
        aB = *(const v16bf*)(Arow + k + 32);
#pragma unroll
        for (int t = 0; t < 4; ++t) bB[t] = *(const v16bf*)(Brow[t] + k + 32);
    }
    // drain: last two chunks
#pragma unroll
    for (int t = 0; t < 4; ++t) acc[t] = wmma_bf16(aA, bA[t], acc[t]);
#pragma unroll
    for (int t = 0; t < 4; ++t) acc[t] = wmma_bf16(aB, bB[t], acc[t]);

#pragma unroll
    for (int t = 0; t < 4; ++t) {
#pragma unroll
        for (int r = 0; r < 8; ++r) {
            const int row = wm * 16 + r + hi * 8;       // C layout: row = r + 8*hi
            const int col = wn * 64 + t * 16 + l16;     // col = lane&15 within n-tile
            const float v = acc[t][r] * scale;
            if (mode == 2) {
                ((float*)out)[(size_t)row * N + col] = v;
            } else if (mode == 1) {
                const int b  = row / Tdim, tt = row % Tdim;
                const int h  = col >> 6,  d  = col & 63;
                ((__bf16*)out)[(((size_t)b * Hheads + h) * Tdim + tt) * 64 + d] = (__bf16)v;
            } else {
                ((__bf16*)out)[(size_t)row * N + col] = (__bf16)v;
            }
        }
    }
}

// ---------------------------------------------------------------------------
// RoPE (faithful to source: theta_i = 10000^(i/64), i in [0,32)), in-place bf16
// y: [BH, T, 64]; one thread per rotation pair
// ---------------------------------------------------------------------------
__global__ void rope_kernel(__bf16* __restrict__ y, int totalRows, int T) {
    int idx = blockIdx.x * blockDim.x + threadIdx.x;
    if (idx >= totalRows * 32) return;
    const int i   = idx & 31;
    const int row = idx >> 5;
    const int t   = row % T;
    __bf16* p = y + (size_t)row * 64 + 2 * i;
    const float theta = __powf(10000.0f, (float)i * (1.0f / 64.0f));
    float s, c;
    sincosf((float)t * theta, &s, &c);
    const float x0 = (float)p[0], x1 = (float)p[1];
    p[0] = (__bf16)(x0 * c - x1 * s);
    p[1] = (__bf16)(x0 * s + x1 * c);
}

// ---------------------------------------------------------------------------
// Causal flash attention. Q,K,V: bf16 [B*H, T, 64] (scale already folded in).
// One wave owns a 16-row query tile; streams keys 32 at a time.
// V tile is staged into LDS row-major via async global->LDS loads (ASYNCcnt
// path), transposed at read time for the B fragments.
// Output: bf16 token-major [B*T, H*64].
// ---------------------------------------------------------------------------
__global__ __launch_bounds__(128)
void flash_attn_kernel(const __bf16* __restrict__ Q, const __bf16* __restrict__ Kt,
                       const __bf16* __restrict__ V, __bf16* __restrict__ O,
                       int T, int H) {
    __shared__ __align__(32) __bf16 Plds[4][16][32];  // P tile, C->A transpose
    __shared__ __align__(32) __bf16 Vrow[4][32][64];  // V tile row-major [key][d]

    const int lane = threadIdx.x & 31;
    const int w    = threadIdx.x >> 5;
    const int gw   = blockIdx.x * 4 + w;
    const int qTiles = T >> 4;
    const int bh = gw / qTiles;
    const int qt = gw % qTiles;
    const int q0 = qt * 16;
    const int l16 = lane & 15, hi = lane >> 4;
    const size_t base = (size_t)bh * T * 64;

    // Q A-fragments: 16 rows x 64 (two k-chunks of 32)
    const __bf16* qrow = Q + base + (size_t)(q0 + l16) * 64 + hi * 16;
    const v16bf aq0 = *(const v16bf*)(qrow);
    const v16bf aq1 = *(const v16bf*)(qrow + 32);

    float m[8], l[8];
    v8f o[4] = {v8f{}, v8f{}, v8f{}, v8f{}};
#pragma unroll
    for (int r = 0; r < 8; ++r) { m[r] = -1e30f; l[r] = 0.0f; }

    const int kend = q0 + 15;
    for (int kb = 0; kb <= kend; kb += 32) {
        // ---- kick off async V staging first so it overlaps S + softmax ----
        {
            const int key = kb + lane;
            const __bf16* vrow = V + base + (size_t)key * 64;
#if HAVE_ASYNC_LDS
#pragma unroll
            for (int j = 0; j < 8; ++j) {
                __builtin_amdgcn_global_load_async_to_lds_b128(
                    (AS1 v4i_gcc*)(vrow + j * 8),
                    (AS3 v4i_gcc*)&Vrow[w][lane][j * 8], 0, 0);
            }
#else
#pragma unroll
            for (int j = 0; j < 8; ++j) {
                bf16x8 ch = *(const bf16x8*)(vrow + j * 8);
                *(bf16x8*)&Vrow[w][lane][j * 8] = ch;
            }
#endif
        }
        // speculative prefetch of next tile's K and V rows
        __builtin_prefetch(Kt + base + (size_t)(kb + 32 + l16) * 64, 0, 3);
        __builtin_prefetch(V  + base + (size_t)(kb + 32 + lane) * 64, 0, 3);

        // ---- S = Q K^T for 32 keys (two 16-wide fragments) ----
        v8f s[2];
#pragma unroll
        for (int f = 0; f < 2; ++f) {
            const int key = kb + f * 16 + l16;
            const __bf16* krow = Kt + base + (size_t)key * 64 + hi * 16;
            v16bf b0 = *(const v16bf*)(krow);
            v16bf b1 = *(const v16bf*)(krow + 32);
            v8f sf = {};
            sf = wmma_bf16(aq0, b0, sf);
            sf = wmma_bf16(aq1, b1, sf);
            s[f] = sf;
        }
        // ---- causal mask (key > query row) ----
#pragma unroll
        for (int f = 0; f < 2; ++f) {
            const int key = kb + f * 16 + l16;
#pragma unroll
            for (int r = 0; r < 8; ++r)
                if (key > q0 + r + hi * 8) s[f][r] = -1e30f;
        }
        // ---- online softmax (reductions across 16-lane halves) ----
        float alpha[8];
#pragma unroll
        for (int r = 0; r < 8; ++r) {
            float tmax = fmaxf(s[0][r], s[1][r]);
#pragma unroll
            for (int off = 1; off < 16; off <<= 1)
                tmax = fmaxf(tmax, __shfl_xor(tmax, off, 32));
            const float mn = fmaxf(m[r], tmax);
            alpha[r] = __expf(m[r] - mn);
            m[r] = mn;
            const float p0 = __expf(s[0][r] - mn);
            const float p1 = __expf(s[1][r] - mn);
            s[0][r] = p0; s[1][r] = p1;
            float rs = p0 + p1;
#pragma unroll
            for (int off = 1; off < 16; off <<= 1)
                rs += __shfl_xor(rs, off, 32);
            l[r] = l[r] * alpha[r] + rs;
        }
        // ---- rescale O accumulators ----
#pragma unroll
        for (int t = 0; t < 4; ++t)
#pragma unroll
            for (int r = 0; r < 8; ++r) o[t][r] *= alpha[r];

        // ---- P (C layout) -> LDS as bf16 for A-fragment reads ----
#pragma unroll
        for (int f = 0; f < 2; ++f)
#pragma unroll
            for (int r = 0; r < 8; ++r)
                Plds[w][r + hi * 8][f * 16 + l16] = (__bf16)s[f][r];

#if HAVE_ASYNC_LDS
#if __has_builtin(__builtin_amdgcn_s_wait_asynccnt)
        __builtin_amdgcn_s_wait_asynccnt(0);
#else
        asm volatile("s_wait_asynccnt 0" ::: "memory");
#endif
#endif
        asm volatile("s_wait_dscnt 0" ::: "memory");
        __builtin_amdgcn_wave_barrier();

        // ---- O += P @ V ----
        const v16bf ap = *(const v16bf*)&Plds[w][l16][hi * 16];
#pragma unroll
        for (int t = 0; t < 4; ++t) {
            const int d = t * 16 + l16;
            v16bf bv;
#pragma unroll
            for (int j = 0; j < 16; ++j)
                bv[j] = Vrow[w][hi * 16 + j][d];   // transpose at read
            o[t] = wmma_bf16(ap, bv, o[t]);
        }
        __builtin_amdgcn_wave_barrier();
    }

    // ---- normalize + store token-major [B*T, H*64] ----
    const int b = bh / H, h = bh % H;
#pragma unroll
    for (int r = 0; r < 8; ++r) {
        const float inv = 1.0f / l[r];
        const int tok = q0 + r + hi * 8;
#pragma unroll
        for (int t = 0; t < 4; ++t) {
            const int d = t * 16 + l16;
            O[(size_t)(b * T + tok) * (H * 64) + h * 64 + d] = (__bf16)(o[t][r] * inv);
        }
    }
}

// ---------------------------------------------------------------------------
extern "C" void kernel_launch(void* const* d_in, const int* in_sizes, int n_in,
                              void* d_out, int out_size, void* d_ws, size_t ws_size,
                              hipStream_t stream) {
    (void)in_sizes; (void)n_in; (void)out_size; (void)ws_size;
    constexpr int B = 2, T = 2048, D = 1024, H = 16;
    constexpr int BT = B * T;                 // 4096
    const float qk_scale = 0.3535533905932738f;  // 64^-0.25

    const float* x  = (const float*)d_in[0];
    const float* wq = (const float*)d_in[1];
    const float* wk = (const float*)d_in[2];
    const float* wv = (const float*)d_in[3];
    const float* wo = (const float*)d_in[4];

    char* w = (char*)d_ws;
    __bf16* xb  = (__bf16*)(w);                          // 8 MB
    __bf16* wqb = (__bf16*)(w + (8u  << 20));            // 2 MB
    __bf16* wkb = (__bf16*)(w + (10u << 20));
    __bf16* wvb = (__bf16*)(w + (12u << 20));
    __bf16* wob = (__bf16*)(w + (14u << 20));
    __bf16* Qb  = (__bf16*)(w + (16u << 20));            // 8 MB  [B,H,T,64]
    __bf16* Kb  = (__bf16*)(w + (24u << 20));            // 8 MB
    __bf16* Vb  = (__bf16*)(w + (32u << 20));            // 8 MB
    __bf16* ab  = (__bf16*)(w + (40u << 20));            // 8 MB  attn out [B*T, D]

    // 1) convert inputs to bf16
    cvt_f32_bf16_kernel<<<(BT * D + 255) / 256, 256, 0, stream>>>(x,  xb,  BT * D);
    cvt_f32_bf16_kernel<<<(D * D + 255) / 256, 256, 0, stream>>>(wq, wqb, D * D);
    cvt_f32_bf16_kernel<<<(D * D + 255) / 256, 256, 0, stream>>>(wk, wkb, D * D);
    cvt_f32_bf16_kernel<<<(D * D + 255) / 256, 256, 0, stream>>>(wv, wvb, D * D);
    cvt_f32_bf16_kernel<<<(D * D + 255) / 256, 256, 0, stream>>>(wo, wob, D * D);

    // 2) QKV projections (head-major epilogue, qk_scale folded into Q,K)
    const int gemmWaves  = (BT / 16) * (D / 64);         // 4096 waves
    const int gemmBlocks = gemmWaves / 8;                // 256 thr = 8 waves
    gemm_bf16_kernel<<<gemmBlocks, 256, 0, stream>>>(xb, wqb, Qb, BT, D, D, 1, qk_scale, T, H);
    gemm_bf16_kernel<<<gemmBlocks, 256, 0, stream>>>(xb, wkb, Kb, BT, D, D, 1, qk_scale, T, H);
    gemm_bf16_kernel<<<gemmBlocks, 256, 0, stream>>>(xb, wvb, Vb, BT, D, D, 1, 1.0f,     T, H);

    // 3) RoPE on Q and K
    const int ropeThreads = B * H * T * 32;              // one per pair
    rope_kernel<<<(ropeThreads + 255) / 256, 256, 0, stream>>>(Qb, B * H * T, T);
    rope_kernel<<<(ropeThreads + 255) / 256, 256, 0, stream>>>(Kb, B * H * T, T);

    // 4) causal flash attention -> token-major bf16
    const int flashWaves  = B * H * (T / 16);            // 4096 waves
    flash_attn_kernel<<<flashWaves / 4, 128, 0, stream>>>(Qb, Kb, Vb, ab, T, H);

    // 5) output projection -> fp32 d_out
    gemm_bf16_kernel<<<gemmBlocks, 256, 0, stream>>>(ab, wob, (float*)d_out,
                                                     BT, D, D, 2, 1.0f, T, H);
}
